// Head_41025527612002
// MI455X (gfx1250) — matrix operand order repaired and checked by reading, not compiled, not run on hardware
//
#include <hip/hip_runtime.h>

typedef __bf16 bf16;
typedef __attribute__((ext_vector_type(16))) __bf16 v16bf;
typedef __attribute__((ext_vector_type(8)))  __bf16 v8bf;
typedef __attribute__((ext_vector_type(8)))  float  v8f;
typedef __attribute__((ext_vector_type(4)))  float  v4f;

#define B_ 8
#define T_ 2048
#define E_ 1024
#define D_ 128
#define MTOT (B_ * T_)              // 16384 rows
#define SCALE 11.313708498984761f   // sqrt(128): reference does scores * d**0.5

static __device__ __forceinline__ v16bf cat16(v8bf lo, v8bf hi) {
  return __builtin_shufflevector(lo, hi, 0,1,2,3,4,5,6,7,8,9,10,11,12,13,14,15);
}

static __device__ __forceinline__ v8f wmma_bf16(v16bf a, v16bf b, v8f c) {
  return __builtin_amdgcn_wmma_f32_16x16x32_bf16(false, a, false, b,
                                                 (short)0, c, false, false);
}

// ---------------------------------------------------------------------------
// Kernel 1: fused QKV projection.  x[16384,1024]f32 @ {wq,wk,wv}[1024,128]f32
// -> Q,K,V bf16.  x is read ONCE; each A-fragment feeds 12 WMMAs.
// grid = 256 blocks (64 rows each), block = 256 (8 waves).
// wave = (msub 0..3, nhalf 0..1): 16 rows x 64 cols x 3 matrices.
// Double-buffered LDS weight slabs + register-double-buffered x fragments:
// one __syncthreads() per k-step, global loads in flight under WMMA chain.
// ---------------------------------------------------------------------------
__global__ __launch_bounds__(256) void qkv_proj_kernel(
    const float* __restrict__ x,
    const float* __restrict__ wq, const float* __restrict__ wk,
    const float* __restrict__ wv,
    bf16* __restrict__ Q, bf16* __restrict__ K, bf16* __restrict__ V)
{
  const int tid  = threadIdx.x;
  const int wave = tid >> 5;
  const int lane = tid & 31;
  const int ln   = lane & 15;
  const int hi   = lane >> 4;
  const int msub  = wave >> 1;        // 0..3
  const int nhalf = wave & 1;         // 0..1

  const int mrow = blockIdx.x * 64 + msub * 16;
  const int ncol = nhalf * 64;

  __shared__ bf16 wT[2][3][128][40];  // [buf][mat][n][k] transposed, padded

  const float* wmat[3] = {wq, wk, wv};
  bf16*        omat[3] = {Q, K, V};

  // staging mapping: thread -> (k row, 16-col span) of each 32x128 w slab
  const int skk = tid >> 3;           // 0..31
  const int sn0 = (tid & 7) * 16;     // 0..112

  v4f sreg[3][4];
  #pragma unroll
  for (int m = 0; m < 3; ++m) {
    const float* src = wmat[m] + (size_t)skk * D_ + sn0;
    #pragma unroll
    for (int j = 0; j < 4; ++j) sreg[m][j] = ((const v4f*)src)[j];
  }

  // x prefetch registers (chunk 0)
  const float* xrow = x + (size_t)(mrow + ln) * E_ + (hi ? 8 : 0);
  v4f xs0 = ((const v4f*)xrow)[0];
  v4f xs1 = ((const v4f*)xrow)[1];
  v4f xs2 = ((const v4f*)(xrow + 16))[0];
  v4f xs3 = ((const v4f*)(xrow + 16))[1];

  v8f acc[3][4];
  #pragma unroll
  for (int m = 0; m < 3; ++m)
    #pragma unroll
    for (int t = 0; t < 4; ++t)
      #pragma unroll
      for (int i = 0; i < 8; ++i) acc[m][t][i] = 0.0f;

  for (int kc = 0; kc < E_ / 32; ++kc) {
    const int buf = kc & 1;

    // ---- store staged weight slab (transposed) into LDS buffer `buf`
    #pragma unroll
    for (int m = 0; m < 3; ++m)
      #pragma unroll
      for (int j = 0; j < 4; ++j)
        #pragma unroll
        for (int i = 0; i < 4; ++i)
          wT[buf][m][sn0 + j * 4 + i][skk] = (bf16)sreg[m][j][i];

    // ---- build A fragment from current x registers
    v16bf a;
    #pragma unroll
    for (int i = 0; i < 4; ++i) {
      a[i]      = (bf16)xs0[i];  a[i + 4]  = (bf16)xs1[i];
      a[i + 8]  = (bf16)xs2[i];  a[i + 12] = (bf16)xs3[i];
    }

    // ---- prefetch next chunk (global loads fly under the WMMA chain)
    if (kc + 1 < E_ / 32) {
      const int kb2 = (kc + 1) * 32;
      #pragma unroll
      for (int m = 0; m < 3; ++m) {
        const float* src = wmat[m] + (size_t)(kb2 + skk) * D_ + sn0;
        #pragma unroll
        for (int j = 0; j < 4; ++j) sreg[m][j] = ((const v4f*)src)[j];
      }
      const float* xp = xrow + kb2;
      xs0 = ((const v4f*)xp)[0];
      xs1 = ((const v4f*)xp)[1];
      xs2 = ((const v4f*)(xp + 16))[0];
      xs3 = ((const v4f*)(xp + 16))[1];
    }

    __syncthreads();   // single barrier per k-step (double-buffered LDS)

    #pragma unroll
    for (int m = 0; m < 3; ++m) {
      // load 4 B-fragments first, then 4 back-to-back WMMAs
      v16bf bfr[4];
      #pragma unroll
      for (int nt = 0; nt < 4; ++nt) {
        const bf16* bp = &wT[buf][m][ncol + nt * 16 + ln][hi ? 8 : 0];
        bfr[nt] = cat16(*(const v8bf*)bp, *(const v8bf*)(bp + 16));
      }
      #pragma unroll
      for (int nt = 0; nt < 4; ++nt)
        acc[m][nt] = wmma_bf16(a, bfr[nt], acc[m][nt]);
    }
  }

  #pragma unroll
  for (int m = 0; m < 3; ++m)
    #pragma unroll
    for (int nt = 0; nt < 4; ++nt)
      #pragma unroll
      for (int r = 0; r < 8; ++r)
        omat[m][(size_t)(mrow + r + hi * 8) * D_ + ncol + nt * 16 + ln] =
            (bf16)acc[m][nt][r];
}

// ---------------------------------------------------------------------------
// Kernel 2: causal flash attention over bf16 Q,K,V (f32 accumulate/softmax)
// grid = (T/128, B), block = 256 (8 waves). Wave owns 16 query rows x D=128.
// Double-buffered K/V slabs: one __syncthreads() per 32-key chunk.
// ---------------------------------------------------------------------------
__global__ __launch_bounds__(256) void attn_kernel(
    const bf16* __restrict__ Q, const bf16* __restrict__ K,
    const bf16* __restrict__ V, float* __restrict__ out)
{
  const int tid  = threadIdx.x;
  const int wave = tid >> 5;
  const int lane = tid & 31;
  const int ln   = lane & 15;
  const int hi   = lane >> 4;

  const int b  = blockIdx.y;
  const int q0 = blockIdx.x * 128 + wave * 16;
  const size_t base = (size_t)b * T_ * D_;

  __shared__ bf16 Klds[2][32][136];    // [buf][s][d], padded
  __shared__ bf16 Vlds[2][128][40];    // [buf][d][s] transposed, padded
  __shared__ bf16 Plds[8][16][40];     // per-wave P tile [m][s]

  // Preload Q A-fragments for 4 d-chunks of 32
  v16bf qa[4];
  {
    const bf16* qrow = Q + base + (size_t)(q0 + ln) * D_;
    #pragma unroll
    for (int kd = 0; kd < 4; ++kd) {
      const bf16* p = qrow + kd * 32 + (hi ? 8 : 0);
      qa[kd] = cat16(*(const v8bf*)p, *(const v8bf*)(p + 16));
    }
  }

  v8f acc[8];
  float rowm[8], rowl[8];
  #pragma unroll
  for (int t = 0; t < 8; ++t) {
    rowm[t] = -1e30f; rowl[t] = 0.0f;
    #pragma unroll
    for (int i = 0; i < 8; ++i) acc[t][i] = 0.0f;
  }

  const int nchunks = blockIdx.x * 4 + 4;   // keys <= block's max query row

  // staging maps
  const int ks  = tid >> 3, kd0 = (tid & 7) * 16;   // K slab copy
  const int vs  = tid & 31, vd0 = (tid >> 5) * 16;  // V slab transpose

  // prologue: stage chunk 0 into registers
  v8bf kr0, kr1, vr0, vr1;
  {
    const bf16* ksrc = K + base + (size_t)ks * D_ + kd0;
    kr0 = *(const v8bf*)ksrc;  kr1 = *(const v8bf*)(ksrc + 8);
    const bf16* vsrc = V + base + (size_t)vs * D_ + vd0;
    vr0 = *(const v8bf*)vsrc;  vr1 = *(const v8bf*)(vsrc + 8);
  }

  for (int sc = 0; sc < nchunks; ++sc) {
    const int sb  = sc * 32;
    const int buf = sc & 1;

    // ---- commit staged registers to LDS buffer `buf`
    *(v8bf*)&Klds[buf][ks][kd0]     = kr0;
    *(v8bf*)&Klds[buf][ks][kd0 + 8] = kr1;
    #pragma unroll
    for (int i = 0; i < 8; ++i) {
      Vlds[buf][vd0 + i][vs]     = vr0[i];
      Vlds[buf][vd0 + 8 + i][vs] = vr1[i];
    }

    // ---- prefetch next chunk
    if (sc + 1 < nchunks) {
      const int sb2 = sb + 32;
      const bf16* ksrc = K + base + (size_t)(sb2 + ks) * D_ + kd0;
      kr0 = *(const v8bf*)ksrc;  kr1 = *(const v8bf*)(ksrc + 8);
      const bf16* vsrc = V + base + (size_t)(sb2 + vs) * D_ + vd0;
      vr0 = *(const v8bf*)vsrc;  vr1 = *(const v8bf*)(vsrc + 8);
    }

    __syncthreads();   // single barrier per chunk (double-buffered LDS)

    // ---- S = Q @ K^T for two 16-key subtiles
    v8f s0, s1;
    #pragma unroll
    for (int i = 0; i < 8; ++i) { s0[i] = 0.0f; s1[i] = 0.0f; }
    #pragma unroll
    for (int kd = 0; kd < 4; ++kd) {
      const bf16* kp0 = &Klds[buf][ln][kd * 32 + (hi ? 8 : 0)];
      const bf16* kp1 = &Klds[buf][16 + ln][kd * 32 + (hi ? 8 : 0)];
      v16bf b0 = cat16(*(const v8bf*)kp0, *(const v8bf*)(kp0 + 16));
      v16bf b1 = cat16(*(const v8bf*)kp1, *(const v8bf*)(kp1 + 16));
      s0 = wmma_bf16(qa[kd], b0, s0);
      s1 = wmma_bf16(qa[kd], b1, s1);
    }

    // ---- scale, causal mask, online softmax update
    const int col0 = sb + ln, col1 = col0 + 16;
    #pragma unroll
    for (int r = 0; r < 8; ++r) {
      const int row = q0 + r + hi * 8;
      float a0 = s0[r] * SCALE, a1 = s1[r] * SCALE;
      if (col0 > row) a0 = -1e30f;
      if (col1 > row) a1 = -1e30f;
      float v = fmaxf(a0, a1);
      v = fmaxf(v, __shfl_xor(v, 1));
      v = fmaxf(v, __shfl_xor(v, 2));
      v = fmaxf(v, __shfl_xor(v, 4));
      v = fmaxf(v, __shfl_xor(v, 8));
      const float mnew  = fmaxf(rowm[r], v);
      const float alpha = __expf(rowm[r] - mnew);
      rowm[r] = mnew;
      const float p0 = __expf(a0 - mnew);
      const float p1 = __expf(a1 - mnew);
      float sum = p0 + p1;
      sum += __shfl_xor(sum, 1);
      sum += __shfl_xor(sum, 2);
      sum += __shfl_xor(sum, 4);
      sum += __shfl_xor(sum, 8);
      rowl[r] = rowl[r] * alpha + sum;
      #pragma unroll
      for (int nt = 0; nt < 8; ++nt) acc[nt][r] *= alpha;
      Plds[wave][r + hi * 8][ln]      = (bf16)p0;
      Plds[wave][r + hi * 8][ln + 16] = (bf16)p1;
    }

    // ---- P C-layout -> A-layout via per-wave LDS round trip
    const bf16* pp = &Plds[wave][ln][hi ? 8 : 0];
    v16bf pa = cat16(*(const v8bf*)pp, *(const v8bf*)(pp + 16));

    // ---- acc += P @ V : 8 d-tiles, fragments loaded in bursts of 4
    #pragma unroll
    for (int g = 0; g < 2; ++g) {
      v16bf vb[4];
      #pragma unroll
      for (int j = 0; j < 4; ++j) {
        const bf16* vp = &Vlds[buf][(g * 4 + j) * 16 + ln][hi ? 8 : 0];
        vb[j] = cat16(*(const v8bf*)vp, *(const v8bf*)(vp + 16));
      }
      #pragma unroll
      for (int j = 0; j < 4; ++j)
        acc[g * 4 + j] = wmma_bf16(pa, vb[j], acc[g * 4 + j]);
    }
  }

  // ---- epilogue: normalize and store f32
  #pragma unroll
  for (int nt = 0; nt < 8; ++nt)
    #pragma unroll
    for (int r = 0; r < 8; ++r) {
      const int row = q0 + r + hi * 8;
      out[((size_t)b * T_ + row) * D_ + nt * 16 + ln] = acc[nt][r] / rowl[r];
    }
}

extern "C" void kernel_launch(void* const* d_in, const int* in_sizes, int n_in,
                              void* d_out, int out_size, void* d_ws, size_t ws_size,
                              hipStream_t stream) {
  const float* x  = (const float*)d_in[0];
  const float* wq = (const float*)d_in[1];
  const float* wk = (const float*)d_in[2];
  const float* wv = (const float*)d_in[3];

  bf16* Q = (bf16*)d_ws;                       // 16384*128 bf16 = 4 MB each
  bf16* K = Q + (size_t)MTOT * D_;
  bf16* V = K + (size_t)MTOT * D_;

  dim3 g1(MTOT / 64);
  qkv_proj_kernel<<<g1, 256, 0, stream>>>(x, wq, wk, wv, Q, K, V);

  dim3 g2(T_ / 128, B_);
  attn_kernel<<<g2, 256, 0, stream>>>(Q, K, V, (float*)d_out);
}
